// MultiProductCNN_with_Attention_and_GAT_25237227831838
// MI455X (gfx1250) — compile-verified, matrix operand order checked by reading, not run on hardware
//
#include <hip/hip_runtime.h>

typedef __bf16 bf16_t;
typedef bf16_t v16bf __attribute__((ext_vector_type(16)));
typedef bf16_t v8bf  __attribute__((ext_vector_type(8)));
typedef float  v8f   __attribute__((ext_vector_type(8)));

#define M_MAIN 131072        // B*P*S = 8*512*32
#define EDGE_N 8192
#define E2     8704          // edges + self loops
#define LDSA   40            // LDS row stride for BK=32 tiles (bank-conflict pad)
#define LDK    72            // LDS row stride for BK=64 tiles (bank-conflict pad)

// ---------------------------------------------------------------------------
// helpers
// ---------------------------------------------------------------------------
__device__ __forceinline__ v8f v8f_zero() {
  v8f z = {0.f,0.f,0.f,0.f,0.f,0.f,0.f,0.f};
  return z;
}
__device__ __forceinline__ v8bf v8bf_zero() {
  v8bf z = {(bf16_t)0.f,(bf16_t)0.f,(bf16_t)0.f,(bf16_t)0.f,
            (bf16_t)0.f,(bf16_t)0.f,(bf16_t)0.f,(bf16_t)0.f};
  return z;
}
// build a 16-element bf16 WMMA fragment from two aligned 16-byte chunks
__device__ __forceinline__ v16bf frag_from(const bf16_t* p0, const bf16_t* p1) {
  v8bf lo = *(const v8bf*)p0;
  v8bf hi = *(const v8bf*)p1;
  return __builtin_shufflevector(lo, hi, 0,1,2,3,4,5,6,7,8,9,10,11,12,13,14,15);
}
__device__ __forceinline__ v8f wmma_bf16(v16bf a, v16bf b, v8f c) {
  return __builtin_amdgcn_wmma_f32_16x16x32_bf16(false, a, false, b, (short)0, c, false, false);
}
__device__ __forceinline__ void atomicMaxF(float* addr, float val) {
  unsigned int* ai = (unsigned int*)addr;
  unsigned int old = *ai;
  while (true) {
    float f = __uint_as_float(old);
    if (f >= val) break;
    unsigned int assumed = old;
    old = atomicCAS(ai, assumed, __float_as_uint(val));
    if (old == assumed) break;
  }
}

// epilogue: if scale: c = acc*scale[n]+shift[n]; else c = acc + bias[n]; opt relu
__device__ __forceinline__ void gemm_epilogue(
    v8f acc[2][2], int bm, int bn, int wr, int wc, int lane,
    int M, int N, const float* bias, const float* scale, const float* shift,
    int relu, float* outF, bf16_t* outH)
{
  const int mo = (lane & 16) ? 8 : 0;   // C/D layout: lanes 16-31 hold M=r+8
  #pragma unroll
  for (int i = 0; i < 2; ++i) {
    #pragma unroll
    for (int j = 0; j < 2; ++j) {
      const int n = bn + wc * 32 + j * 16 + (lane & 15);
      float sc = 1.f, sh = 0.f;
      if (scale)      { sc = scale[n]; sh = shift[n]; }
      else if (bias)  { sh = bias[n]; }
      #pragma unroll
      for (int r = 0; r < 8; ++r) {
        const int m = bm + wr * 32 + i * 16 + r + mo;
        if (m < M) {
          float c = acc[i][j][r] * sc + sh;
          if (relu) c = fmaxf(c, 0.f);
          if (outF) outF[(size_t)m * N + n] = c;
          if (outH) outH[(size_t)m * N + n] = (bf16_t)c;
        }
      }
    }
  }
}

// ---------------------------------------------------------------------------
// fast bf16 WMMA GEMM:  out[M,N] = A[M,K] @ Wt[N,K]^T
// requires M % 128 == 0, N % 64 == 0, K % 64 == 0
// double-buffered LDS, BK=64 -> 8 WMMAs per barrier per wave
// ---------------------------------------------------------------------------
__global__ __launch_bounds__(256) void k_gemm_fast(
    const bf16_t* __restrict__ A, const bf16_t* __restrict__ Wt,
    int M, int N, int K,
    const float* __restrict__ bias, const float* __restrict__ scale,
    const float* __restrict__ shift, int relu,
    float* __restrict__ outF, bf16_t* __restrict__ outH)
{
  __shared__ __attribute__((aligned(16))) bf16_t As[2][128 * LDK];
  __shared__ __attribute__((aligned(16))) bf16_t Bs[2][64 * LDK];
  const int tid  = threadIdx.x;
  const int lane = tid & 31;
  const int wave = tid >> 5;
  const int wr   = wave & 3;
  const int wc   = wave >> 2;
  const int bm   = blockIdx.x * 128;
  const int bn   = blockIdx.y * 64;

  const int arow = tid >> 1;           // A tile: 128 rows x 64, 32 elems/thread
  const int acol = (tid & 1) * 32;
  const int brow = tid >> 2;           // B tile: 64 rows x 64, 16 elems/thread
  const int bcol = (tid & 3) * 16;

  const bf16_t* aptr = A  + (size_t)(bm + arow) * K + acol;
  const bf16_t* bptr = Wt + (size_t)(bn + brow) * K + bcol;

  v8f acc[2][2];
  #pragma unroll
  for (int i = 0; i < 2; ++i)
    #pragma unroll
    for (int j = 0; j < 2; ++j) acc[i][j] = v8f_zero();

  const int ml  = wr * 32 + (lane & 15);
  const int nl  = wc * 32 + (lane & 15);
  const int klo = (lane & 16) ? 8 : 0;

  v8bf ar0 = *(const v8bf*)(aptr + 0);
  v8bf ar1 = *(const v8bf*)(aptr + 8);
  v8bf ar2 = *(const v8bf*)(aptr + 16);
  v8bf ar3 = *(const v8bf*)(aptr + 24);
  v8bf br0 = *(const v8bf*)(bptr + 0);
  v8bf br1 = *(const v8bf*)(bptr + 8);
  {
    bf16_t* as = &As[0][arow * LDK + acol];
    *(v8bf*)(as + 0) = ar0;  *(v8bf*)(as + 8)  = ar1;
    *(v8bf*)(as + 16) = ar2; *(v8bf*)(as + 24) = ar3;
    bf16_t* bs = &Bs[0][brow * LDK + bcol];
    *(v8bf*)(bs + 0) = br0;  *(v8bf*)(bs + 8)  = br1;
  }
  __syncthreads();

  const int nsteps = K >> 6;
  for (int s = 0; s < nsteps; ++s) {
    const int cur = s & 1, nxt = cur ^ 1;
    const bool more = (s + 1) < nsteps;
    if (more) {
      const bf16_t* ap = aptr + (size_t)(s + 1) * 64;
      ar0 = *(const v8bf*)(ap + 0);  ar1 = *(const v8bf*)(ap + 8);
      ar2 = *(const v8bf*)(ap + 16); ar3 = *(const v8bf*)(ap + 24);
      const bf16_t* bp = bptr + (size_t)(s + 1) * 64;
      br0 = *(const v8bf*)(bp + 0);  br1 = *(const v8bf*)(bp + 8);
    }
    const bf16_t* asb = As[cur];
    const bf16_t* bsb = Bs[cur];
    #pragma unroll
    for (int kk = 0; kk < 64; kk += 32) {
      v16bf af0 = frag_from(&asb[ml * LDK + kk + klo],        &asb[ml * LDK + kk + klo + 16]);
      v16bf af1 = frag_from(&asb[(ml + 16) * LDK + kk + klo], &asb[(ml + 16) * LDK + kk + klo + 16]);
      v16bf bf0 = frag_from(&bsb[nl * LDK + kk + klo],        &bsb[nl * LDK + kk + klo + 16]);
      v16bf bf1 = frag_from(&bsb[(nl + 16) * LDK + kk + klo], &bsb[(nl + 16) * LDK + kk + klo + 16]);
      acc[0][0] = wmma_bf16(af0, bf0, acc[0][0]);
      acc[0][1] = wmma_bf16(af0, bf1, acc[0][1]);
      acc[1][0] = wmma_bf16(af1, bf0, acc[1][0]);
      acc[1][1] = wmma_bf16(af1, bf1, acc[1][1]);
    }
    if (more) {
      bf16_t* as = &As[nxt][arow * LDK + acol];
      *(v8bf*)(as + 0) = ar0;  *(v8bf*)(as + 8)  = ar1;
      *(v8bf*)(as + 16) = ar2; *(v8bf*)(as + 24) = ar3;
      bf16_t* bs = &Bs[nxt][brow * LDK + bcol];
      *(v8bf*)(bs + 0) = br0;  *(v8bf*)(bs + 8)  = br1;
    }
    __syncthreads();
  }

  gemm_epilogue(acc, bm, bn, wr, wc, lane, M, N, bias, scale, shift, relu, outF, outH);
}

// ---------------------------------------------------------------------------
// small-M bf16 WMMA GEMM (M arbitrary, K % 32 == 0, N % 64 == 0)
// OOB A rows are clamped (their garbage stays in unstored C rows)
// ---------------------------------------------------------------------------
__global__ __launch_bounds__(256) void k_gemm_small(
    const bf16_t* __restrict__ A, const bf16_t* __restrict__ Wt,
    int M, int N, int K,
    const float* __restrict__ bias, const float* __restrict__ scale,
    const float* __restrict__ shift, int relu,
    float* __restrict__ outF, bf16_t* __restrict__ outH)
{
  __shared__ __attribute__((aligned(16))) bf16_t As[128 * LDSA];
  __shared__ __attribute__((aligned(16))) bf16_t Bs[64 * LDSA];
  const int tid  = threadIdx.x;
  const int lane = tid & 31;
  const int wave = tid >> 5;
  const int wr   = wave & 3;
  const int wc   = wave >> 2;
  const int bm   = blockIdx.x * 128;
  const int bn   = blockIdx.y * 64;

  const int arow = tid >> 1, ahalf = tid & 1;
  const int brow = tid >> 2, bq    = tid & 3;

  const int arc = (bm + arow < M) ? (bm + arow) : (M - 1);   // clamp, no branch in loop
  const bf16_t* aptr = A  + (size_t)arc * K + ahalf * 16;
  const bf16_t* bptr = Wt + (size_t)(bn + brow) * K + bq * 8;

  v8f acc[2][2];
  #pragma unroll
  for (int i = 0; i < 2; ++i)
    #pragma unroll
    for (int j = 0; j < 2; ++j) acc[i][j] = v8f_zero();

  const int ml  = wr * 32 + (lane & 15);
  const int nl  = wc * 32 + (lane & 15);
  const int klo = (lane & 16) ? 8 : 0;

  for (int k0 = 0; k0 < K; k0 += 32) {
    v8bf a0 = *(const v8bf*)(aptr + k0);
    v8bf a1 = *(const v8bf*)(aptr + k0 + 8);
    v8bf bv = *(const v8bf*)(bptr + k0);

    __syncthreads();
    *(v8bf*)&As[arow * LDSA + ahalf * 16]     = a0;
    *(v8bf*)&As[arow * LDSA + ahalf * 16 + 8] = a1;
    *(v8bf*)&Bs[brow * LDSA + bq * 8]         = bv;
    __syncthreads();

    v16bf af0 = frag_from(&As[ml * LDSA + klo],        &As[ml * LDSA + klo + 16]);
    v16bf af1 = frag_from(&As[(ml + 16) * LDSA + klo], &As[(ml + 16) * LDSA + klo + 16]);
    v16bf bf0 = frag_from(&Bs[nl * LDSA + klo],        &Bs[nl * LDSA + klo + 16]);
    v16bf bf1 = frag_from(&Bs[(nl + 16) * LDSA + klo], &Bs[(nl + 16) * LDSA + klo + 16]);

    acc[0][0] = wmma_bf16(af0, bf0, acc[0][0]);
    acc[0][1] = wmma_bf16(af0, bf1, acc[0][1]);
    acc[1][0] = wmma_bf16(af1, bf0, acc[1][0]);
    acc[1][1] = wmma_bf16(af1, bf1, acc[1][1]);
  }

  gemm_epilogue(acc, bm, bn, wr, wc, lane, M, N, bias, scale, shift, relu, outF, outH);
}

// ---------------------------------------------------------------------------
// conv1: C_in=1, 3x3, pad 1 + BN + ReLU, output NHWC bf16 [8,512,32,64]
// ---------------------------------------------------------------------------
__global__ __launch_bounds__(256) void k_conv1(
    const float* __restrict__ x, const float* __restrict__ w,
    const float* __restrict__ cb, const float* __restrict__ g,
    const float* __restrict__ bb, const float* __restrict__ m,
    const float* __restrict__ vv, bf16_t* __restrict__ out)
{
  int idx = blockIdx.x * 256 + threadIdx.x;
  if (idx >= 8 * 512 * 32 * 64) return;
  int c  = idx & 63;
  int xx = (idx >> 6) & 31;
  int y  = (idx >> 11) & 511;
  int b  = idx >> 20;
  const float* xb = x + (size_t)b * 16384;
  float acc = 0.f;
  #pragma unroll
  for (int dy = 0; dy < 3; ++dy) {
    int yy = y + dy - 1;
    if (yy < 0 || yy >= 512) continue;
    #pragma unroll
    for (int dx = 0; dx < 3; ++dx) {
      int x2 = xx + dx - 1;
      if (x2 < 0 || x2 >= 32) continue;
      acc += xb[yy * 32 + x2] * w[c * 9 + dy * 3 + dx];
    }
  }
  float sc = g[c] * rsqrtf(vv[c] + 1e-5f);
  float r  = fmaxf((acc + cb[c]) * sc + (bb[c] - m[c] * sc), 0.f);
  out[idx] = (bf16_t)r;
}

// im2col (NHWC bf16 -> [M, taps*C] bf16), K-order = (kh*KW+kw)*C + c
__global__ __launch_bounds__(256) void k_im2col(
    const bf16_t* __restrict__ in, bf16_t* __restrict__ col,
    int C, int KH, int KW, int padH, int padW)
{
  int taps = KH * KW;
  long long tot = (long long)M_MAIN * taps;
  long long id  = (long long)blockIdx.x * 256 + threadIdx.x;
  if (id >= tot) return;
  int tap = (int)(id % taps);
  int m   = (int)(id / taps);
  int kw = tap % KW, kh = tap / KW;
  int xx = m & 31, y = (m >> 5) & 511, b = m >> 14;
  int sy = y + kh - padH, sx = xx + kw - padW;
  bf16_t* dst = col + (size_t)m * (taps * C) + (size_t)tap * C;
  if (sy >= 0 && sy < 512 && sx >= 0 && sx < 32) {
    const bf16_t* src = in + (size_t)((b * 512 + sy) * 32 + sx) * C;
    for (int c = 0; c < C; c += 8) *(v8bf*)(dst + c) = *(const v8bf*)(src + c);
  } else {
    v8bf z = v8bf_zero();
    for (int c = 0; c < C; c += 8) *(v8bf*)(dst + c) = z;
  }
}

// repack conv weight [O][C][KH][KW] f32 -> [O][(kh*KW+kw)*C + c] bf16
__global__ __launch_bounds__(256) void k_repack(
    const float* __restrict__ w, bf16_t* __restrict__ o,
    int O, int C, int KH, int KW)
{
  int id = blockIdx.x * 256 + threadIdx.x;
  if (id >= O * C * KH * KW) return;
  int kw = id % KW; int t = id / KW;
  int kh = t % KH;  t /= KH;
  int c  = t % C;   int oo = t / C;
  o[(size_t)oo * (C * KH * KW) + (size_t)(kh * KW + kw) * C + c] = (bf16_t)w[id];
}

__global__ __launch_bounds__(256) void k_cvt(const float* __restrict__ in,
                                             bf16_t* __restrict__ out, int n)
{
  int i = blockIdx.x * 256 + threadIdx.x;
  if (i < n) out[i] = (bf16_t)in[i];
}

__global__ __launch_bounds__(256) void k_fill(float* p, float v, int n)
{
  int i = blockIdx.x * 256 + threadIdx.x;
  if (i < n) p[i] = v;
}

// fold BN (and optional preceding bias) into per-channel scale/shift
__global__ void k_prep_bn(const float* g, const float* b, const float* m,
                          const float* v, const float* cb,
                          float* sc, float* sh, int n)
{
  int i = blockIdx.x * 64 + threadIdx.x;
  if (i >= n) return;
  float s = g[i] * rsqrtf(v[i] + 1e-5f);
  sc[i] = s;
  sh[i] = (cb ? cb[i] * s : 0.f) + b[i] - m[i] * s;
}

// ---------------------------------------------------------------------------
// SE block
// ---------------------------------------------------------------------------
__global__ __launch_bounds__(256) void k_semean(const float* __restrict__ h3,
                                                float* __restrict__ mean)
{
  int b = blockIdx.x, chunk = blockIdx.y, c = threadIdx.x;
  const float* p = h3 + ((size_t)b * 16384 + (size_t)chunk * 512) * 256 + c;
  float a = 0.f;
  for (int i = 0; i < 512; ++i) a += p[(size_t)i * 256];
  atomicAdd(&mean[b * 256 + c], a);
}

__global__ __launch_bounds__(256) void k_sefc(
    const float* __restrict__ mean, const float* __restrict__ w1,
    const float* __restrict__ b1, const float* __restrict__ w2,
    const float* __restrict__ b2, float* __restrict__ se)
{
  __shared__ float sm[256];
  __shared__ float s1[16];
  int b = blockIdx.x, t = threadIdx.x;
  sm[t] = mean[b * 256 + t] * (1.f / 16384.f);
  __syncthreads();
  if (t < 16) {
    float a = b1[t];
    for (int c = 0; c < 256; ++c) a += w1[t * 256 + c] * sm[c];
    s1[t] = fmaxf(a, 0.f);
  }
  __syncthreads();
  float a = b2[t];
  for (int j = 0; j < 16; ++j) a += w2[t * 16 + j] * s1[j];
  se[b * 256 + t] = 1.f / (1.f + expf(-a));
}

__global__ __launch_bounds__(256) void k_apply_se(
    const float* __restrict__ h3, const float* __restrict__ se,
    float* __restrict__ tF, bf16_t* __restrict__ tH)
{
  int idx = blockIdx.x * 256 + threadIdx.x;   // 33554432 total
  int c = idx & 255;
  int b = idx >> 22;
  float v = h3[idx] * se[b * 256 + c];
  tF[idx] = v;
  tH[idx] = (bf16_t)v;
}

// ---------------------------------------------------------------------------
// attention: per (seq-index, head) wave; S=32, HD=32; WMMA for QK^T and PV
// ---------------------------------------------------------------------------
__global__ __launch_bounds__(128) void k_attn(const bf16_t* __restrict__ qkv,
                                              bf16_t* __restrict__ ctx)
{
  __shared__ __attribute__((aligned(16))) float  sc[4][32][33];
  __shared__ __attribute__((aligned(16))) bf16_t at[4][32][LDSA];
  __shared__ __attribute__((aligned(16))) bf16_t vt[4][32][LDSA];
  const int nidx = blockIdx.x;
  const int wave = threadIdx.x >> 5;
  const int lane = threadIdx.x & 31;
  const int h    = blockIdx.y * 4 + wave;
  const bf16_t* base = qkv + (size_t)nidx * 32 * 768;
  const int mr  = lane & 15;
  const int klo = (lane & 16) ? 8 : 0;
  const int mo  = (lane & 16) ? 8 : 0;

  // scores = q @ k^T / sqrt(HD)
  v16bf qf[2], kf[2];
  #pragma unroll
  for (int i = 0; i < 2; ++i) {
    const bf16_t* qp = base + (size_t)(i * 16 + mr) * 768 + h * 32;
    qf[i] = frag_from(qp + klo, qp + klo + 16);
    const bf16_t* kp = qp + 256;
    kf[i] = frag_from(kp + klo, kp + klo + 16);
  }
  v8f s[2][2];
  #pragma unroll
  for (int i = 0; i < 2; ++i)
    #pragma unroll
    for (int j = 0; j < 2; ++j) s[i][j] = v8f_zero();
  s[0][0] = wmma_bf16(qf[0], kf[0], s[0][0]);
  s[0][1] = wmma_bf16(qf[0], kf[1], s[0][1]);
  s[1][0] = wmma_bf16(qf[1], kf[0], s[1][0]);
  s[1][1] = wmma_bf16(qf[1], kf[1], s[1][1]);

  const float scl = 0.17677669529663687f;   // 1/sqrt(32)
  #pragma unroll
  for (int i = 0; i < 2; ++i)
    #pragma unroll
    for (int j = 0; j < 2; ++j)
      #pragma unroll
      for (int r = 0; r < 8; ++r)
        sc[wave][i * 16 + r + mo][j * 16 + (lane & 15)] = s[i][j][r] * scl;

  // v^T into LDS: vt[d][s_key]
  const bf16_t* vp = base + (size_t)lane * 768 + 512 + h * 32;
  for (int d = 0; d < 32; ++d) vt[wave][d][lane] = vp[d];
  __syncthreads();

  // softmax per query row (lane = row)
  float rowv[32];
  float mx = -1e30f;
  #pragma unroll
  for (int k2 = 0; k2 < 32; ++k2) { rowv[k2] = sc[wave][lane][k2]; mx = fmaxf(mx, rowv[k2]); }
  float sum = 0.f;
  #pragma unroll
  for (int k2 = 0; k2 < 32; ++k2) { rowv[k2] = expf(rowv[k2] - mx); sum += rowv[k2]; }
  float inv = 1.f / sum;
  #pragma unroll
  for (int k2 = 0; k2 < 32; ++k2) at[wave][lane][k2] = (bf16_t)(rowv[k2] * inv);
  __syncthreads();

  // ctx = attn @ v
  v16bf af[2], bfv[2];
  #pragma unroll
  for (int i = 0; i < 2; ++i) {
    af[i]  = frag_from(&at[wave][i * 16 + mr][klo], &at[wave][i * 16 + mr][klo + 16]);
    bfv[i] = frag_from(&vt[wave][i * 16 + mr][klo], &vt[wave][i * 16 + mr][klo + 16]);
  }
  v8f c2[2][2];
  #pragma unroll
  for (int i = 0; i < 2; ++i)
    #pragma unroll
    for (int j = 0; j < 2; ++j) c2[i][j] = v8f_zero();
  c2[0][0] = wmma_bf16(af[0], bfv[0], c2[0][0]);
  c2[0][1] = wmma_bf16(af[0], bfv[1], c2[0][1]);
  c2[1][0] = wmma_bf16(af[1], bfv[0], c2[1][0]);
  c2[1][1] = wmma_bf16(af[1], bfv[1], c2[1][1]);

  #pragma unroll
  for (int i = 0; i < 2; ++i)
    #pragma unroll
    for (int j = 0; j < 2; ++j)
      #pragma unroll
      for (int r = 0; r < 8; ++r)
        ctx[((size_t)nidx * 32 + i * 16 + r + mo) * 256 + h * 32 + j * 16 + (lane & 15)]
            = (bf16_t)c2[i][j][r];
}

// ---------------------------------------------------------------------------
// layernorm with residual: y = LN(x + r) * g + b  (D = 256, one row per wave)
// ---------------------------------------------------------------------------
__global__ __launch_bounds__(256) void k_ln(
    const float* __restrict__ x, const float* __restrict__ r,
    const float* __restrict__ g, const float* __restrict__ b,
    float* __restrict__ outF, bf16_t* __restrict__ outH, int rows)
{
  int wave = threadIdx.x >> 5, lane = threadIdx.x & 31;
  int row = blockIdx.x * 8 + wave;
  if (row >= rows) return;
  const float* xp = x + (size_t)row * 256;
  const float* rp = r ? r + (size_t)row * 256 : nullptr;
  float v[8];
  float s = 0.f;
  #pragma unroll
  for (int t = 0; t < 8; ++t) {
    float a = xp[t * 32 + lane];
    if (rp) a += rp[t * 32 + lane];
    v[t] = a; s += a;
  }
  #pragma unroll
  for (int o = 16; o >= 1; o >>= 1) s += __shfl_xor(s, o, 32);
  float mu = s * (1.f / 256.f);
  float q = 0.f;
  #pragma unroll
  for (int t = 0; t < 8; ++t) { float d = v[t] - mu; q += d * d; }
  #pragma unroll
  for (int o = 16; o >= 1; o >>= 1) q += __shfl_xor(q, o, 32);
  float rstd = rsqrtf(q * (1.f / 256.f) + 1e-5f);
  #pragma unroll
  for (int t = 0; t < 8; ++t) {
    int c = t * 32 + lane;
    float y = (v[t] - mu) * rstd * g[c] + b[c];
    if (outF) outF[(size_t)row * 256 + c] = y;
    if (outH) outH[(size_t)row * 256 + c] = (bf16_t)y;
  }
}

// ---------------------------------------------------------------------------
// GAT pieces
// ---------------------------------------------------------------------------
__global__ __launch_bounds__(256) void k_gat_alpha(
    const float* __restrict__ xh, const float* __restrict__ asrc,
    const float* __restrict__ adst, float* __restrict__ As_, float* __restrict__ Ad_)
{
  int id = blockIdx.x * 256 + threadIdx.x;   // 8*512*4
  if (id >= 16384) return;
  int h = id & 3, n = id >> 2;
  const float* xp = xh + ((size_t)n * 4 + h) * 64;
  float s = 0.f, d = 0.f;
  for (int c = 0; c < 64; ++c) { float v = xp[c]; s += v * asrc[h * 64 + c]; d += v * adst[h * 64 + c]; }
  As_[id] = s; Ad_[id] = d;
}

__global__ __launch_bounds__(256) void k_gat_e1(
    const int* __restrict__ ei, const float* __restrict__ As_,
    const float* __restrict__ Ad_, float* __restrict__ ev, float* __restrict__ emax)
{
  int id = blockIdx.x * 256 + threadIdx.x;   // 8*E2*4
  if (id >= 8 * E2 * 4) return;
  int h = id & 3, e = (id >> 2) % E2, b = id / (E2 * 4);
  int src, dst;
  if (e < EDGE_N) { src = ei[e]; dst = ei[EDGE_N + e]; } else { src = dst = e - EDGE_N; }
  float v = As_[(b * 512 + src) * 4 + h] + Ad_[(b * 512 + dst) * 4 + h];
  v = v > 0.f ? v : 0.2f * v;
  ev[id] = v;
  atomicMaxF(&emax[(b * 512 + dst) * 4 + h], v);
}

__global__ __launch_bounds__(256) void k_gat_e2(
    const int* __restrict__ ei, float* __restrict__ ev,
    const float* __restrict__ emax, float* __restrict__ denom)
{
  int id = blockIdx.x * 256 + threadIdx.x;
  if (id >= 8 * E2 * 4) return;
  int h = id & 3, e = (id >> 2) % E2, b = id / (E2 * 4);
  int dst = (e < EDGE_N) ? ei[EDGE_N + e] : (e - EDGE_N);
  float p = expf(ev[id] - emax[(b * 512 + dst) * 4 + h]);
  ev[id] = p;
  atomicAdd(&denom[(b * 512 + dst) * 4 + h], p);
}

__global__ __launch_bounds__(256) void k_gat_e3(
    const int* __restrict__ ei, const float* __restrict__ ev,
    const float* __restrict__ denom, const float* __restrict__ xh,
    float* __restrict__ outg)
{
  int id = blockIdx.x * 256 + threadIdx.x;   // 8*E2*4*16
  if (id >= 8 * E2 * 64) return;
  int cg = id & 15;
  int h  = (id >> 4) & 3;
  int e  = (id >> 6) % E2;
  int b  = id / (E2 * 64);
  int src, dst;
  if (e < EDGE_N) { src = ei[e]; dst = ei[EDGE_N + e]; } else { src = dst = e - EDGE_N; }
  float alpha = ev[((size_t)(b * E2 + e)) * 4 + h]
              / (denom[(b * 512 + dst) * 4 + h] + 1e-16f);
  const float* xp = xh + ((size_t)(b * 512 + src) * 4 + h) * 64 + cg * 4;
  float* op = outg + ((size_t)(b * 512 + dst) * 4 + h) * 64 + cg * 4;
  atomicAdd(op + 0, xp[0] * alpha);
  atomicAdd(op + 1, xp[1] * alpha);
  atomicAdd(op + 2, xp[2] * alpha);
  atomicAdd(op + 3, xp[3] * alpha);
}

__global__ __launch_bounds__(256) void k_gat_fin(
    const float* __restrict__ outg, const float* __restrict__ gb,
    bf16_t* __restrict__ flat)
{
  int id = blockIdx.x * 256 + threadIdx.x;   // 8*512*256
  if (id >= 1048576) return;
  int j = id & 255;
  flat[id] = (bf16_t)(outg[id] + gb[j]);
}

// ---------------------------------------------------------------------------
// launcher
// ---------------------------------------------------------------------------
extern "C" void kernel_launch(void* const* d_in, const int* in_sizes, int n_in,
                              void* d_out, int out_size, void* d_ws, size_t ws_size,
                              hipStream_t stream)
{
  const float* xin     = (const float*)d_in[0];
  const int*   eidx    = (const int*)  d_in[1];
  const float* conv1_w = (const float*)d_in[2];
  const float* conv1_b = (const float*)d_in[3];
  const float* bn1_g = (const float*)d_in[4],  *bn1_b = (const float*)d_in[5];
  const float* bn1_m = (const float*)d_in[6],  *bn1_v = (const float*)d_in[7];
  const float* conv2_w = (const float*)d_in[8];
  const float* conv2_b = (const float*)d_in[9];
  const float* bn2_g = (const float*)d_in[10], *bn2_b = (const float*)d_in[11];
  const float* bn2_m = (const float*)d_in[12], *bn2_v = (const float*)d_in[13];
  const float* conv3_w = (const float*)d_in[14];
  const float* conv3_b = (const float*)d_in[15];
  const float* bn3_g = (const float*)d_in[16], *bn3_b = (const float*)d_in[17];
  const float* bn3_m = (const float*)d_in[18], *bn3_v = (const float*)d_in[19];
  const float* se_w1 = (const float*)d_in[20], *se_b1 = (const float*)d_in[21];
  const float* se_w2 = (const float*)d_in[22], *se_b2 = (const float*)d_in[23];
  const float* attn_in_w  = (const float*)d_in[24], *attn_in_b  = (const float*)d_in[25];
  const float* attn_out_w = (const float*)d_in[26], *attn_out_b = (const float*)d_in[27];
  const float* ln1_g = (const float*)d_in[28], *ln1_b = (const float*)d_in[29];
  const float* ffn1_w = (const float*)d_in[30], *ffn1_b = (const float*)d_in[31];
  const float* ffn2_w = (const float*)d_in[32], *ffn2_b = (const float*)d_in[33];
  const float* ln2_g = (const float*)d_in[34], *ln2_b = (const float*)d_in[35];
  const float* gat_w  = (const float*)d_in[36];
  const float* att_src = (const float*)d_in[37], *att_dst = (const float*)d_in[38];
  const float* gat_b  = (const float*)d_in[39];
  const float* fc1_w = (const float*)d_in[40], *fc1_b = (const float*)d_in[41];
  const float* bh1_g = (const float*)d_in[42], *bh1_b = (const float*)d_in[43];
  const float* bh1_m = (const float*)d_in[44], *bh1_v = (const float*)d_in[45];
  const float* fc2_w = (const float*)d_in[46], *fc2_b = (const float*)d_in[47];
  const float* bh2_g = (const float*)d_in[48], *bh2_b = (const float*)d_in[49];
  const float* bh2_m = (const float*)d_in[50], *bh2_v = (const float*)d_in[51];
  const float* fc3_w = (const float*)d_in[52], *fc3_b = (const float*)d_in[53];
  float* out = (float*)d_out;
  (void)in_sizes; (void)n_in; (void)out_size; (void)ws_size;

  char* wsb = (char*)d_ws;
  size_t off = 0;
  auto alloc = [&](size_t bytes) -> void* {
    void* p = wsb + off;
    off += (bytes + 255) & ~((size_t)255);
    return p;
  };

  bf16_t* h1b   = (bf16_t*)alloc((size_t)8 * 512 * 32 * 64 * 2);
  bf16_t* col   = (bf16_t*)alloc((size_t)M_MAIN * 2688 * 2);
  bf16_t* h2b   = (bf16_t*)alloc((size_t)M_MAIN * 128 * 2);
  float*  h3f   = (float*) alloc((size_t)M_MAIN * 256 * 4);
  float*  semn  = (float*) alloc(2048 * 4);
  float*  sev   = (float*) alloc(2048 * 4);
  float*  tF    = (float*) alloc((size_t)M_MAIN * 256 * 4);
  bf16_t* tH    = (bf16_t*)alloc((size_t)M_MAIN * 256 * 2);
  bf16_t* qkvH  = (bf16_t*)alloc((size_t)M_MAIN * 768 * 2);
  bf16_t* ctxH  = (bf16_t*)alloc((size_t)M_MAIN * 256 * 2);
  float*  aoF   = (float*) alloc((size_t)M_MAIN * 256 * 4);
  float*  x1F   = (float*) alloc((size_t)M_MAIN * 256 * 4);
  bf16_t* x1H   = (bf16_t*)alloc((size_t)M_MAIN * 256 * 2);
  bf16_t* ffnH  = (bf16_t*)alloc((size_t)M_MAIN * 1024 * 2);
  float*  f2F   = (float*) alloc((size_t)M_MAIN * 256 * 4);
  bf16_t* x2H   = (bf16_t*)alloc((size_t)M_MAIN * 256 * 2);
  float*  xh    = (float*) alloc((size_t)4096 * 256 * 4);
  float*  alS   = (float*) alloc(16384 * 4);
  float*  alD   = (float*) alloc(16384 * 4);
  float*  emax  = (float*) alloc(16384 * 4);
  float*  denom = (float*) alloc(16384 * 4);
  float*  ev    = (float*) alloc((size_t)8 * E2 * 4 * 4);
  float*  outg  = (float*) alloc((size_t)8 * 512 * 256 * 4);
  bf16_t* flatH = (bf16_t*)alloc((size_t)8 * 131072 * 2);
  bf16_t* z1H   = (bf16_t*)alloc(8 * 128 * 2);
  bf16_t* z2H   = (bf16_t*)alloc(8 * 64 * 2);
  bf16_t* Wc2   = (bf16_t*)alloc((size_t)122880 * 2);
  bf16_t* Wc3   = (bf16_t*)alloc((size_t)688128 * 2);
  bf16_t* Wqkv  = (bf16_t*)alloc((size_t)196608 * 2);
  bf16_t* Wout  = (bf16_t*)alloc((size_t)65536 * 2);
  bf16_t* Wf1   = (bf16_t*)alloc((size_t)262144 * 2);
  bf16_t* Wf2   = (bf16_t*)alloc((size_t)262144 * 2);
  bf16_t* Wgat  = (bf16_t*)alloc((size_t)2097152 * 2);
  bf16_t* Wfc1  = (bf16_t*)alloc((size_t)16777216 * 2);
  bf16_t* Wfc2  = (bf16_t*)alloc((size_t)8192 * 2);
  bf16_t* Wfc3  = (bf16_t*)alloc((size_t)32768 * 2);
  float* bn2s = (float*)alloc(128 * 4); float* bn2t = (float*)alloc(128 * 4);
  float* bn3s = (float*)alloc(256 * 4); float* bn3t = (float*)alloc(256 * 4);
  float* bh1s = (float*)alloc(128 * 4); float* bh1t = (float*)alloc(128 * 4);
  float* bh2s = (float*)alloc(64 * 4);  float* bh2t = (float*)alloc(64 * 4);

  auto gemm = [&](const bf16_t* A, const bf16_t* Wt, int M, int N, int K,
                  const float* bias, const float* scale, const float* shift,
                  int relu, float* oF, bf16_t* oH) {
    dim3 g((M + 127) / 128, N / 64);
    if ((M & 127) == 0 && (K & 63) == 0)
      k_gemm_fast<<<g, dim3(256), 0, stream>>>(A, Wt, M, N, K, bias, scale, shift, relu, oF, oH);
    else
      k_gemm_small<<<g, dim3(256), 0, stream>>>(A, Wt, M, N, K, bias, scale, shift, relu, oF, oH);
  };

  // ---- weight prep ----
  k_prep_bn<<<dim3(2), dim3(64), 0, stream>>>(bn2_g, bn2_b, bn2_m, bn2_v, conv2_b, bn2s, bn2t, 128);
  k_prep_bn<<<dim3(4), dim3(64), 0, stream>>>(bn3_g, bn3_b, bn3_m, bn3_v, conv3_b, bn3s, bn3t, 256);
  k_prep_bn<<<dim3(2), dim3(64), 0, stream>>>(bh1_g, bh1_b, bh1_m, bh1_v, fc1_b, bh1s, bh1t, 128);
  k_prep_bn<<<dim3(1), dim3(64), 0, stream>>>(bh2_g, bh2_b, bh2_m, bh2_v, fc2_b, bh2s, bh2t, 64);
  k_repack<<<dim3(480),  dim3(256), 0, stream>>>(conv2_w, Wc2, 128, 64, 3, 5);
  k_repack<<<dim3(2688), dim3(256), 0, stream>>>(conv3_w, Wc3, 256, 128, 3, 7);
  k_cvt<<<dim3(768),   dim3(256), 0, stream>>>(attn_in_w,  Wqkv, 196608);
  k_cvt<<<dim3(256),   dim3(256), 0, stream>>>(attn_out_w, Wout, 65536);
  k_cvt<<<dim3(1024),  dim3(256), 0, stream>>>(ffn1_w, Wf1, 262144);
  k_cvt<<<dim3(1024),  dim3(256), 0, stream>>>(ffn2_w, Wf2, 262144);
  k_cvt<<<dim3(8192),  dim3(256), 0, stream>>>(gat_w,  Wgat, 2097152);
  k_cvt<<<dim3(65536), dim3(256), 0, stream>>>(fc1_w,  Wfc1, 16777216);
  k_cvt<<<dim3(32),    dim3(256), 0, stream>>>(fc2_w,  Wfc2, 8192);
  k_cvt<<<dim3(128),   dim3(256), 0, stream>>>(fc3_w,  Wfc3, 32768);

  // ---- CNN backbone (NHWC) ----
  k_conv1<<<dim3(32768), dim3(256), 0, stream>>>(xin, conv1_w, conv1_b,
                                                 bn1_g, bn1_b, bn1_m, bn1_v, h1b);
  k_im2col<<<dim3(7680), dim3(256), 0, stream>>>(h1b, col, 64, 3, 5, 1, 2);
  gemm(col, Wc2, M_MAIN, 128, 960, nullptr, bn2s, bn2t, 1, nullptr, h2b);
  k_im2col<<<dim3(10752), dim3(256), 0, stream>>>(h2b, col, 128, 3, 7, 1, 3);
  gemm(col, Wc3, M_MAIN, 256, 2688, nullptr, bn3s, bn3t, 1, h3f, nullptr);

  // ---- SE ----
  k_fill<<<dim3(8), dim3(256), 0, stream>>>(semn, 0.f, 2048);
  k_semean<<<dim3(8, 32), dim3(256), 0, stream>>>(h3f, semn);
  k_sefc<<<dim3(8), dim3(256), 0, stream>>>(semn, se_w1, se_b1, se_w2, se_b2, sev);
  k_apply_se<<<dim3(131072), dim3(256), 0, stream>>>(h3f, sev, tF, tH);

  // ---- transformer ----
  gemm(tH, Wqkv, M_MAIN, 768, 256, attn_in_b, nullptr, nullptr, 0, nullptr, qkvH);
  k_attn<<<dim3(4096, 2), dim3(128), 0, stream>>>(qkvH, ctxH);
  gemm(ctxH, Wout, M_MAIN, 256, 256, attn_out_b, nullptr, nullptr, 0, aoF, nullptr);
  k_ln<<<dim3(16384), dim3(256), 0, stream>>>(tF, aoF, ln1_g, ln1_b, x1F, x1H, M_MAIN);
  gemm(x1H, Wf1, M_MAIN, 1024, 256, ffn1_b, nullptr, nullptr, 1, nullptr, ffnH);
  gemm(ffnH, Wf2, M_MAIN, 256, 1024, ffn2_b, nullptr, nullptr, 0, f2F, nullptr);
  k_ln<<<dim3(16384), dim3(256), 0, stream>>>(x1F, f2F, ln2_g, ln2_b, nullptr, x2H, M_MAIN);

  // ---- GAT ----
  gemm(x2H, Wgat, 4096, 256, 8192, nullptr, nullptr, nullptr, 0, xh, nullptr);
  k_gat_alpha<<<dim3(64), dim3(256), 0, stream>>>(xh, att_src, att_dst, alS, alD);
  k_fill<<<dim3(64),   dim3(256), 0, stream>>>(emax, -1e30f, 16384);
  k_fill<<<dim3(64),   dim3(256), 0, stream>>>(denom, 0.f, 16384);
  k_fill<<<dim3(4096), dim3(256), 0, stream>>>(outg, 0.f, 1048576);
  k_gat_e1<<<dim3(1088),  dim3(256), 0, stream>>>(eidx, alS, alD, ev, emax);
  k_gat_e2<<<dim3(1088),  dim3(256), 0, stream>>>(eidx, ev, emax, denom);
  k_gat_e3<<<dim3(17408), dim3(256), 0, stream>>>(eidx, ev, denom, xh, outg);
  k_gat_fin<<<dim3(4096), dim3(256), 0, stream>>>(outg, gat_b, flatH);

  // ---- head ----
  gemm(flatH, Wfc1, 8, 128, 131072, nullptr, bh1s, bh1t, 1, nullptr, z1H);
  gemm(z1H,   Wfc2, 8, 64, 128,     nullptr, bh2s, bh2t, 1, nullptr, z2H);
  gemm(z2H,   Wfc3, 8, 512, 64,     fc3_b,   nullptr, nullptr, 0, out, nullptr);
}
